// TripleAdaptiveQuantizerV10_89781996356282
// MI455X (gfx1250) — matrix operand order repaired and checked by reading, not compiled, not run on hardware
//
#include <hip/hip_runtime.h>
#include <math.h>

typedef __attribute__((ext_vector_type(16))) _Float16 v16h;
typedef __attribute__((ext_vector_type(8)))  float    v8f;

#define NBLK   16384          // B*L
#define OFF_V  0
#define OFF_H  1048576
#define OFF_Y  9437184
#define OFF_BT 9961472
#define OFF_BY 10223616
#define OFF_BH 10485760
#define OFF_BD 10747904

// ---------------- weight packing into WMMA B-matrix layout ----------------
// B layout (16-bit, 32x16): lane l element e -> K = (e<8?0:16)+(l<16?0:8)+e%8, N = l%16
// chunks: 0..3  = layer1 (Ncol chunk c): val = (K<9)? W1[(16c+N)*9 + K] : 0
//         4..11 = layer2 (ck*4+cn):      val = W2[(16cn+N)*64 + 32ck + K]
//         12..13= layer3 (ck):           val = (N<15)? W3[N*64 + 32ck + K] : 0
__global__ void pack_weights_kernel(const float* __restrict__ W1,
                                    const float* __restrict__ W2,
                                    const float* __restrict__ W3,
                                    _Float16* __restrict__ wp) {
    int t = blockIdx.x * blockDim.x + threadIdx.x;   // over 14*32*16 = 7168
    if (t >= 14 * 32 * 16) return;
    int e = t & 15;
    int lane = (t >> 4) & 31;
    int chunk = t >> 9;
    int kk = (e < 8) ? (((lane & 16) ? 8 : 0) + e) : (16 + ((lane & 16) ? 8 : 0) + (e - 8));
    int nl = lane & 15;
    float val = 0.0f;
    if (chunk < 4) {
        int n = chunk * 16 + nl;
        if (kk < 9) val = W1[n * 9 + kk];
    } else if (chunk < 12) {
        int c = chunk - 4;
        int ck = c >> 2, cn = c & 3;
        val = W2[(cn * 16 + nl) * 64 + ck * 32 + kk];
    } else {
        int ck = chunk - 12;
        if (nl < 15) val = W3[nl * 64 + ck * 32 + kk];
    }
    wp[t] = (_Float16)val;
}

// ---------------- helpers ----------------
__device__ __forceinline__ float gumbelf(float u) {
    return -logf(-logf(u + 1e-20f) + 1e-20f);
}

struct QP { float s, qp, qn; int j; };

__device__ __forceinline__ QP make_qp(int j, float s1, float s2, float s3, float s4) {
    QP q; q.j = j;
    float s = s1;
    s = (j == 2) ? s2 : s;
    s = (j == 3) ? s3 : s;
    s = (j == 4) ? s4 : s;
    q.s = s;
    int sh = (j >= 1) ? (4 * j - 1) : 1;
    int qpi = (1 << sh) - 1;
    q.qp = (float)qpi;
    q.qn = -(float)(qpi + 1);
    return q;
}

__device__ __forceinline__ float fq(float x, const QP& q) {
    float xc = fminf(fmaxf(x / q.s, q.qn), q.qp);
    float r = rintf(xc) * q.s;             // round half-to-even STE, times scale
    return (q.j == 0) ? 0.0f : r;
}

// ---------------- fused kernel: one wave32 per (b,l) tile ----------------
__global__ void __launch_bounds__(32)
taq_fused_kernel(const float* __restrict__ v,   const float* __restrict__ H,
                 const float* __restrict__ y,   const float* __restrict__ snr,
                 const float* __restrict__ u_y, const float* __restrict__ u_H,
                 const float* __restrict__ u_d,
                 const float* __restrict__ b1,  const float* __restrict__ b2,
                 const float* __restrict__ b3,
                 const float* __restrict__ s4p, const float* __restrict__ s8p,
                 const float* __restrict__ s12p,const float* __restrict__ s16p,
                 const _Float16* __restrict__ wpack,
                 float* __restrict__ out) {
    const int blk  = blockIdx.x;      // b*L + l
    const int lane = threadIdx.x;     // 0..31
    const int r    = lane & 15;
    const int hi   = lane >> 4;

    __shared__ float sh[16][72];      // activation bounce buffer (padded)
    __shared__ float sl[16][16];      // logits tile
    __shared__ int   s_jH[16];
    __shared__ int   s_jd[16];
    __shared__ int   s_jy;

    const float sc1 = *s4p, sc2 = *s8p, sc3 = *s12p, sc4 = *s16p;

    // ---- phase 1: features ----------------------------------------------
    const float* Hb = H + (size_t)blk * 512;
    float hp = 0.0f;
    const int jbase = hi * 8;
#pragma unroll
    for (int j = 0; j < 8; ++j) {
        float2 t = *(const float2*)(Hb + (size_t)(((jbase + j) * 16 + r) * 2));
        hp = fmaf(t.x, t.x, hp);
        hp = fmaf(t.y, t.y, hp);
    }
    hp += __shfl_xor(hp, 16, 32);                 // full column sum (both halves)
    float hs = hp;
    hs += __shfl_xor(hs, 1, 32);
    hs += __shfl_xor(hs, 2, 32);
    hs += __shfl_xor(hs, 4, 32);
    hs += __shfl_xor(hs, 8, 32);                  // sum over all 16 rows
    float sir = log1pf(hp / (hs - hp + 1e-10f));

    const float* yb = y + (size_t)blk * 32;
    float yv = yb[lane];
    float yp = yv * yv;
    yp += __shfl_xor(yp, 1, 32);
    yp += __shfl_xor(yp, 2, 32);
    yp += __shfl_xor(yp, 4, 32);
    yp += __shfl_xor(yp, 8, 32);
    yp += __shfl_xor(yp, 16, 32);
    yp = log1pf(yp);

    const float4 v4 = *(const float4*)(v + (size_t)blk * 64 + r * 4);
    float vn = sqrtf(v4.x * v4.x + v4.y * v4.y + v4.z * v4.z + v4.w * v4.w);
    float sn = snr[(size_t)blk * 16 + r];

    float pf[9];
    pf[0] = v4.x; pf[1] = v4.y; pf[2] = v4.z; pf[3] = v4.w;
    pf[4] = sn;   pf[5] = hp;   pf[6] = sir;  pf[7] = yp;  pf[8] = vn;

    // ---- A1 tile (16x32 f16, features 9..31 zero) ------------------------
    // lane<16: e<8 -> K=e (pf[e]); lane>=16: e=0 -> K=8 (pf[8]); rest zero
    v16h a1;
#pragma unroll
    for (int e = 0; e < 8; ++e) {
        float fv = (hi == 0) ? pf[e] : ((e == 0) ? pf[8] : 0.0f);
        a1[e] = (_Float16)fv;
    }
#pragma unroll
    for (int e = 8; e < 16; ++e) a1[e] = (_Float16)0.0f;

    const v16h* wb = (const v16h*)wpack;

    // ---- layer 1: 4 WMMAs ------------------------------------------------
#pragma unroll
    for (int cn = 0; cn < 4; ++cn) {
        v8f acc = {};
        acc = __builtin_amdgcn_wmma_f32_16x16x32_f16(
            false, a1, false, wb[cn * 32 + lane], (short)0, acc, false, false);
        float bias = b1[cn * 16 + r];
#pragma unroll
        for (int t = 0; t < 8; ++t)
            sh[t + hi * 8][cn * 16 + r] = fmaxf(acc[t] + bias, 0.0f);
    }
    __syncthreads();

    // ---- build A2 (two 16x32 f16 chunks) from LDS ------------------------
    v16h a2c[2];
#pragma unroll
    for (int q = 0; q < 2; ++q) {
#pragma unroll
        for (int e = 0; e < 16; ++e) {
            int kk = 32 * q + ((e < 8) ? (hi * 8 + e) : (16 + hi * 8 + (e - 8)));
            a2c[q][e] = (_Float16)sh[r][kk];
        }
    }
    __syncthreads();

    // ---- layer 2: 8 WMMAs ------------------------------------------------
#pragma unroll
    for (int cn = 0; cn < 4; ++cn) {
        v8f acc = {};
        acc = __builtin_amdgcn_wmma_f32_16x16x32_f16(
            false, a2c[0], false, wb[(4 + cn) * 32 + lane], (short)0, acc, false, false);
        acc = __builtin_amdgcn_wmma_f32_16x16x32_f16(
            false, a2c[1], false, wb[(8 + cn) * 32 + lane], (short)0, acc, false, false);
        float bias = b2[cn * 16 + r];
#pragma unroll
        for (int t = 0; t < 8; ++t)
            sh[t + hi * 8][cn * 16 + r] = fmaxf(acc[t] + bias, 0.0f);
    }
    __syncthreads();

    // ---- build A3 from LDS ----------------------------------------------
    v16h a3c[2];
#pragma unroll
    for (int q = 0; q < 2; ++q) {
#pragma unroll
        for (int e = 0; e < 16; ++e) {
            int kk = 32 * q + ((e < 8) ? (hi * 8 + e) : (16 + hi * 8 + (e - 8)));
            a3c[q][e] = (_Float16)sh[r][kk];
        }
    }

    // ---- layer 3: 2 WMMAs -> logits (cols 0..14 valid) -------------------
    v8f lg = {};
    lg = __builtin_amdgcn_wmma_f32_16x16x32_f16(
        false, a3c[0], false, wb[12 * 32 + lane], (short)0, lg, false, false);
    lg = __builtin_amdgcn_wmma_f32_16x16x32_f16(
        false, a3c[1], false, wb[13 * 32 + lane], (short)0, lg, false, false);
    float bias3 = (r < 15) ? b3[r] : 0.0f;
#pragma unroll
    for (int t = 0; t < 8; ++t)
        sl[t + hi * 8][r] = lg[t] + bias3;
    __syncthreads();

    // ---- gumbel argmax selections (all WMMAs done; divergence OK now) ----
    if (lane < 16) {
        const int k = lane;
        const float* uh = u_H + ((size_t)blk * 16 + k) * 5;
        const float* udp = u_d + ((size_t)blk * 16 + k) * 5;
        int jH = 0, jd = 0;
        float bH = -1e30f, bd = -1e30f;
#pragma unroll
        for (int i = 0; i < 5; ++i) {
            float vh = sl[k][5 + i] + gumbelf(uh[i]);
            if (vh > bH) { bH = vh; jH = i; }
            float vd = sl[k][10 + i] + gumbelf(udp[i]);
            if (vd > bd) { bd = vd; jd = i; }
        }
        s_jH[k] = jH;
        s_jd[k] = jd;
    }
    if (lane == 0) {
        int jy = 0;
        float by = -1e30f;
#pragma unroll
        for (int i = 0; i < 5; ++i) {
            float m = 0.0f;
#pragma unroll
            for (int k = 0; k < 16; ++k) m += sl[k][i];
            m = m * (1.0f / 16.0f) + gumbelf(u_y[(size_t)blk * 5 + i]);
            if (m > by) { by = m; jy = i; }
        }
        s_jy = jy;
    }
    __syncthreads();

    // ---- outputs ---------------------------------------------------------
    const int jy = s_jy;

    // y_q: 32 floats per block, one per lane
    {
        QP q = make_qp(jy, sc1, sc2, sc3, sc4);
        out[OFF_Y + (size_t)blk * 32 + lane] = fq(yv, q);
    }

    // H_q: lane covers row r (axis-2), 16 contiguous floats; selection j_H[r]
    {
        QP q = make_qp(s_jH[r], sc1, sc2, sc3, sc4);
        const float* Hin = Hb + r * 32 + hi * 16;
        float* Hout = out + OFF_H + (size_t)blk * 512 + r * 32 + hi * 16;
#pragma unroll
        for (int t = 0; t < 16; t += 4) {
            float4 hv = *(const float4*)(Hin + t);
            float4 o;
            o.x = fq(hv.x, q); o.y = fq(hv.y, q);
            o.z = fq(hv.z, q); o.w = fq(hv.w, q);
            *(float4*)(Hout + t) = o;
        }
    }

    if (lane < 16) {
        const int k = lane;
        // v_q: float4 per row
        QP q = make_qp(s_jd[k], sc1, sc2, sc3, sc4);
        float4 o;
        o.x = fq(v4.x, q); o.y = fq(v4.y, q);
        o.z = fq(v4.z, q); o.w = fq(v4.w, q);
        *(float4*)(out + OFF_V + (size_t)blk * 64 + k * 4) = o;

        // bits: bv[j] = 4j -> bits_y = 8*jy, bits_H = 128*jH, bits_d = 8*jd
        float bby = 8.0f * (float)jy;
        float bbh = 128.0f * (float)s_jH[k];
        float bbd = 8.0f * (float)s_jd[k];
        size_t o16 = (size_t)blk * 16 + k;
        out[OFF_BT + o16] = bby + bbh + bbd;
        out[OFF_BY + o16] = bby;
        out[OFF_BH + o16] = bbh;
        out[OFF_BD + o16] = bbd;
    }
}

extern "C" void kernel_launch(void* const* d_in, const int* in_sizes, int n_in,
                              void* d_out, int out_size, void* d_ws, size_t ws_size,
                              hipStream_t stream) {
    (void)in_sizes; (void)n_in; (void)out_size; (void)ws_size;
    const float* v   = (const float*)d_in[0];
    const float* H   = (const float*)d_in[1];
    const float* y   = (const float*)d_in[2];
    const float* snr = (const float*)d_in[3];
    const float* uy  = (const float*)d_in[4];
    const float* uH  = (const float*)d_in[5];
    const float* ud  = (const float*)d_in[6];
    const float* W1  = (const float*)d_in[7];
    const float* b1  = (const float*)d_in[8];
    const float* W2  = (const float*)d_in[9];
    const float* b2  = (const float*)d_in[10];
    const float* W3  = (const float*)d_in[11];
    const float* b3  = (const float*)d_in[12];
    const float* s4  = (const float*)d_in[13];
    const float* s8  = (const float*)d_in[14];
    const float* s12 = (const float*)d_in[15];
    const float* s16 = (const float*)d_in[16];
    _Float16* wpack = (_Float16*)d_ws;

    pack_weights_kernel<<<28, 256, 0, stream>>>(W1, W2, W3, wpack);
    taq_fused_kernel<<<NBLK, 32, 0, stream>>>(v, H, y, snr, uy, uH, ud,
                                              b1, b2, b3, s4, s8, s12, s16,
                                              wpack, (float*)d_out);
}
